// GSynthesisBlock_9637906612990
// MI455X (gfx1250) — matrix operand order repaired and checked by reading, not compiled, not run on hardware
//
#include <hip/hip_runtime.h>
#include <hip/hip_bf16.h>
#include <math.h>

// ---------------------------------------------------------------------------
// StyleGAN2 synthesis block for MI455X (gfx1250, wave32, WMMA + TDM).
//
//   conv0_up+blur : fused (k_up (*) flip(w0)) -> 4 phase 3x3 convs, bf16 WMMA
//   conv1         : 3x3 conv, bf16 WMMA
// Conv GEMM: workgroup = 256 thr (8 waves), tile 64oc x 128px, K-tile 32.
//   A (weights) tile : Tensor Data Mover (tensor_load_to_lds) with LDS pad
//                      interval 16 dwords / amount 4 dwords -> 80B row stride
//                      (16B aligned, conflict-free 20-word stride mod 64 banks)
//   B (im2col) tile  : channel-last activations -> two b128 loads + two b128
//                      LDS stores per thread per K-step
//   per wave / K-step: 1 A-frag + 4 B-frags -> 4 v_wmma_f32_16x16x32_bf16
// Roofline: ~3e11 FLOP vs ~0.5 GB traffic -> compute bound -> bf16 WMMA path.
// ---------------------------------------------------------------------------

typedef __attribute__((ext_vector_type(16))) __bf16        v16bf;
typedef __attribute__((ext_vector_type(8)))  float         v8f;
typedef __attribute__((ext_vector_type(8)))  unsigned int  v8u;
typedef __attribute__((ext_vector_type(4)))  unsigned int  u32x4;
typedef __attribute__((ext_vector_type(4)))  int           i32x4;
typedef __attribute__((ext_vector_type(8)))  int           i32x8;

#define SQRT2F 1.41421356237309515f
#define LDS_STRIDE 40   // halves per row: 32 payload + 4-dword TDM pad = 80 B

__device__ __forceinline__ unsigned short f2bf(float f) {
  unsigned int u = __float_as_uint(f);
  unsigned int r = u + 0x7FFFu + ((u >> 16) & 1u);   // round-to-nearest-even
  return (unsigned short)(r >> 16);
}
__device__ __forceinline__ float bf2f(unsigned short h) {
  return __uint_as_float(((unsigned int)h) << 16);
}

// ---------------------------------------------------------------------------
// TDM: load a 64-row x 32-halves 2D tile (row stride = strideElems halves)
// into LDS at ldsByteOff, padding LDS by 4 dwords after every 16 dwords so
// the effective LDS row stride is LDS_STRIDE halves.
// D# fields per CDNA5 ISA.
// ---------------------------------------------------------------------------
__device__ __forceinline__ void tdm_load_a(unsigned int ldsByteOff,
                                           unsigned long long gaddr,
                                           unsigned int strideElems)
{
  u32x4 g0;
  g0[0] = 1u;                                   // count=1, user descriptor
  g0[1] = ldsByteOff;                           // lds_addr (bytes)
  g0[2] = (unsigned int)gaddr;                  // global_addr[31:0]
  g0[3] = (unsigned int)((gaddr >> 32) & 0x01FFFFFFu) | (2u << 30); // type=2
  i32x8 g1;
  g1[0] = (int)((1u << 16)      // data_size = 2 bytes
              | (1u << 20)      // pad_enable
              | (3u << 22)      // pad_interval: 16 dwords
              | (3u << 25));    // pad_amount : 4 dwords
  g1[1] = (int)(0x8000u << 16); // tensor_dim0 = 32768 (low16 in bits 63:48)
  g1[2] = (int)(0x8000u << 16); // dim0 hi16 = 0 | tensor_dim1 low16 = 32768
  g1[3] = (int)(32u << 16);     // dim1 hi16 = 0 | tile_dim0 = 32 elems
  g1[4] = 64;                   // tile_dim1 = 64 rows, tile_dim2 = 0
  g1[5] = (int)strideElems;     // tensor_dim0_stride[31:0]
  g1[6] = 0;
  g1[7] = 0;
  i32x4 z4 = {0, 0, 0, 0};
#if __clang_major__ >= 23
  i32x8 z8 = {0, 0, 0, 0, 0, 0, 0, 0};
  __builtin_amdgcn_tensor_load_to_lds(g0, g1, z4, z4, z8, 0);
#else
  __builtin_amdgcn_tensor_load_to_lds(g0, g1, z4, z4, 0);
#endif
}

// ---------------------------------------------------------------------------
// Styles s0,s1,s2 and demod factors d0,d1 (one block per batch sample).
// ---------------------------------------------------------------------------
__global__ void k_style(const float* __restrict__ dl,
                        const float* __restrict__ m0w, const float* __restrict__ m0b,
                        const float* __restrict__ m1w, const float* __restrict__ m1b,
                        const float* __restrict__ m2w, const float* __restrict__ m2b,
                        const float* __restrict__ w0,  const float* __restrict__ w1,
                        float* __restrict__ s0, float* __restrict__ s1,
                        float* __restrict__ s2, float* __restrict__ d0,
                        float* __restrict__ d1)
{
  const int b = blockIdx.x;
  const int t = threadIdx.x;
  __shared__ float sh0[512];
  __shared__ float sh1[256];
  const float inv512 = 0.04419417382415922f;           // 1/sqrt(512)
  const float* dl9  = dl + ((size_t)b * 14 +  9) * 512;
  const float* dl10 = dl + ((size_t)b * 14 + 10) * 512;
  const float* dl11 = dl + ((size_t)b * 14 + 11) * 512;

  for (int i = t; i < 512; i += 256) {
    float a = 0.f;
    const float* wr = m0w + (size_t)i * 512;
    for (int j = 0; j < 512; ++j) a += dl9[j] * wr[j];
    float v = a * inv512 + m0b[i];
    sh0[i] = v;
    s0[b * 512 + i] = v;
  }
  if (t < 256) {
    float a1 = 0.f, a2 = 0.f;
    const float* w1r = m1w + (size_t)t * 512;
    const float* w2r = m2w + (size_t)t * 512;
    for (int j = 0; j < 512; ++j) { a1 += dl10[j] * w1r[j]; a2 += dl11[j] * w2r[j]; }
    float v1 = a1 * inv512 + m1b[t];
    float v2 = a2 * inv512 + m2b[t];
    sh1[t] = v1;
    s1[b * 256 + t] = v1;
    s2[b * 256 + t] = v2;
  }
  __syncthreads();

  if (t < 256) {
    float acc0 = 0.f;
    for (int i = 0; i < 512; ++i) {
      const float* wp = w0 + ((size_t)t * 512 + i) * 9;
      float ws = 0.f;
      for (int j = 0; j < 9; ++j) ws += wp[j] * wp[j];
      acc0 += ws * sh0[i] * sh0[i];
    }
    d0[b * 256 + t] = rsqrtf(acc0 * (1.f / 4608.f) + 1e-8f);

    float acc1 = 0.f;
    for (int i = 0; i < 256; ++i) {
      const float* wp = w1 + ((size_t)t * 256 + i) * 9;
      float ws = 0.f;
      for (int j = 0; j < 9; ++j) ws += wp[j] * wp[j];
      acc1 += ws * sh1[i] * sh1[i];
    }
    d1[b * 256 + t] = rsqrtf(acc1 * (1.f / 2304.f) + 1e-8f);
  }
}

// ---------------------------------------------------------------------------
// Fuse blur into conv0_up: W6 = conv2d(k_up, flip(w0)*sc0) (6x6), split by
// output parity phase (py,px) into 3x3 kernels: a = 2*ty+1-py, b = 2*tx+1-px.
// Layout: Wp0[ph][oc][tap*512 + ic] bf16.
// ---------------------------------------------------------------------------
__global__ void k_pack_w0(const float* __restrict__ w0, unsigned short* __restrict__ Wp0)
{
  int idx = blockIdx.x * blockDim.x + threadIdx.x;     // idx = o*512 + i
  if (idx >= 256 * 512) return;
  const int o = idx >> 9, i = idx & 511;

  float wl[3][3];
  const float* wp = w0 + ((size_t)o * 512 + i) * 9;
  const float sc0 = 0.014731391274719738f;             // 1/sqrt(512*9)
  for (int ky = 0; ky < 3; ++ky)
    for (int kx = 0; kx < 3; ++kx)
      wl[ky][kx] = wp[ky * 3 + kx] * sc0;

  const float kv[4] = {1.f, 3.f, 3.f, 1.f};            // k_up = outer(kv,kv)/16
  for (int ph = 0; ph < 4; ++ph) {
    const int py = ph >> 1, px = ph & 1;
    for (int ty = 0; ty < 3; ++ty)
      for (int tx = 0; tx < 3; ++tx) {
        const int a  = 2 * ty + 1 - py;
        const int bb = 2 * tx + 1 - px;
        float s = 0.f;
        for (int ky = 0; ky < 3; ++ky) {
          const int u = a - ky;
          if (u < 0 || u > 3) continue;
          for (int kx = 0; kx < 3; ++kx) {
            const int v = bb - kx;
            if (v < 0 || v > 3) continue;
            s += kv[u] * kv[v] * wl[2 - ky][2 - kx];   // flipped w0 (transposed conv)
          }
        }
        s *= (1.f / 16.f);
        Wp0[((((size_t)ph * 256 + o) * 9) + ty * 3 + tx) * 512 + i] = f2bf(s);
      }
  }
}

// Wp1[oc][tap*256 + ic] = w1*sc1 bf16
__global__ void k_pack_w1(const float* __restrict__ w1, unsigned short* __restrict__ Wp1)
{
  int idx = blockIdx.x * blockDim.x + threadIdx.x;
  if (idx >= 256 * 2304) return;
  const int o = idx / 2304, r = idx % 2304;
  const int tap = r >> 8, i = r & 255;
  Wp1[(size_t)o * 2304 + tap * 256 + i] =
      f2bf(w1[((size_t)o * 256 + i) * 9 + tap] * (1.f / 48.f));
}

// ---------------------------------------------------------------------------
// xm[b][yx][ic] = x[b][ic][yx] * s0[b][ic]  (NCHW -> channel-last, bf16)
// 32x32 LDS tile transpose: both global read and write coalesced.
// ---------------------------------------------------------------------------
__global__ void k_modx(const float* __restrict__ x, const float* __restrict__ s0,
                       unsigned short* __restrict__ xm)
{
  const int b  = blockIdx.z;
  const int i0 = blockIdx.y * 32;
  const int p0 = blockIdx.x * 32;
  __shared__ float tile[32][33];
  const int t = threadIdx.x;
#pragma unroll
  for (int j = 0; j < 4; ++j) {
    const int e = t + 256 * j, ii = e >> 5, pp = e & 31;
    tile[ii][pp] = x[((size_t)b * 512 + i0 + ii) * 4096 + p0 + pp] * s0[b * 512 + i0 + ii];
  }
  __syncthreads();
#pragma unroll
  for (int j = 0; j < 4; ++j) {
    const int e = t + 256 * j, pp = e >> 5, ii = e & 31;
    xm[((size_t)b * 4096 + p0 + pp) * 512 + i0 + ii] = f2bf(tile[ii][pp]);
  }
}

// ---------------------------------------------------------------------------
// WMMA fragment packing per ISA 7.12.2 (wave32), LDS row stride 40 halves:
//  A 16x32 bf16: row = lane&15; halves k = (v&4?16:0) + (lane>>4)*8 + (v&3)*2
//  B 32x16 bf16: col = lane&15; halves k = (v&4?16:0) + (lane>>4)*8 + (v&3)*2
//  C/D 16x16 f32: VGPR r -> M = r + 8*(lane>>4), N = lane&15
// ---------------------------------------------------------------------------
__device__ __forceinline__ v16bf load_a_frag(const unsigned short* sA, int row0, int lane)
{
  const unsigned short* rp = sA + (row0 + (lane & 15)) * LDS_STRIDE;
  const int hi = (lane >> 4) * 8;
  v8u u;
#pragma unroll
  for (int v = 0; v < 8; ++v) {
    const int k = ((v & 4) ? 16 : 0) + hi + (v & 3) * 2;
    u[v] = *(const unsigned int*)(rp + k);
  }
  return __builtin_bit_cast(v16bf, u);
}
__device__ __forceinline__ v16bf load_b_frag(const unsigned short* sB, int col0, int lane)
{
  const unsigned short* cp = sB + (col0 + (lane & 15)) * LDS_STRIDE + (lane >> 4) * 8;
  v8u u;
#pragma unroll
  for (int v = 0; v < 8; ++v) u[v] = *(const unsigned int*)(cp + ((v & 4) ? 16 : 0) + (v & 3) * 2);
  return __builtin_bit_cast(v16bf, u);
}

// ---------------------------------------------------------------------------
// Tiled 3x3 conv GEMM, activations channel-last [b][y][x][ic] bf16.
// Block tile 64 oc x 128 px (16w x 8h), wave = 16 oc x 64 px (4 WMMA accums).
// MODE 0: conv0 phase -> demod d0, noise0/bias0/lrelu at (2y+py,2x+px),
//         packed b128 bf16 store to hb (channel-last).
// MODE 1: conv1 -> demod d1, noise1/bias1/lrelu; fp32 NCHW (d_out) + packed
//         bf16 channel-last copy for to_rgb.
// ---------------------------------------------------------------------------
template <int MODE>
__global__ __launch_bounds__(256, 2)
void k_conv3x3_wmma(const unsigned short* __restrict__ act,
                    const unsigned short* __restrict__ wgt,
                    const float* __restrict__ demod,
                    const float* __restrict__ noise,
                    const float* __restrict__ bias,
                    const float* __restrict__ nsPtr,
                    float* __restrict__ outF,
                    unsigned short* __restrict__ outH,
                    int IC, int Hin, int Win, int phase)
{
  const int b      = blockIdx.z;
  const int oc0    = blockIdx.y * 64;
  const int xtiles = Win >> 4;
  const int x0     = (blockIdx.x % xtiles) * 16;
  const int y0     = (blockIdx.x / xtiles) * 8;
  const int t      = threadIdx.x;
  const int lane   = t & 31;
  const int wv     = t >> 5;
  const int om     = wv >> 1;            // 16-oc subtile of this wave
  const int pnb    = (wv & 1) * 4;       // first of four 16-px subtiles
  const int K9     = 9 * IC;

  __shared__ __align__(16) unsigned short sA[64 * LDS_STRIDE];  // TDM dest
  __shared__ __align__(16) unsigned short sB[128 * LDS_STRIDE];

  // touch sA so its address escapes and loads can't be folded (TDM writes it)
  {
    unsigned int z = 0;
#pragma unroll
    for (int j = 0; j < 10; ++j) sA[t * 10 + j] = (unsigned short)z;
  }

  v8f c[4];
#pragma unroll
  for (int i = 0; i < 4; ++i)
#pragma unroll
    for (int r = 0; r < 8; ++r) c[i][r] = 0.f;

  const unsigned short* actb = act + (size_t)b * Hin * Win * IC;
  const unsigned int sA_lds = (unsigned int)(size_t)&sA[0];  // flat->LDS offset

  // per-thread B staging coords: 2 chunks of 8 halves (one b128 each)
  const int p0c = t >> 2, ch0 = t & 3;                 // e = t
  const int p1c = (t + 256) >> 2, ch1 = (t + 256) & 3; // e = t + 256

  for (int tap = 0; tap < 9; ++tap) {
    const int dy = tap / 3 - 1, dx = tap % 3 - 1;
    for (int ic0 = 0; ic0 < IC; ic0 += 32) {
      // A tile via Tensor Data Mover (one wave issues; TENSORcnt tracked)
      if (wv == 0) {
        const unsigned long long ga = (unsigned long long)(size_t)wgt +
            2ull * ((size_t)oc0 * K9 + (size_t)tap * IC + ic0);
        tdm_load_a(sA_lds, ga, (unsigned int)K9);
      }
      // B tile: im2col, channel-last -> b128 load + b128 LDS store per chunk
#pragma unroll
      for (int j = 0; j < 2; ++j) {
        const int p  = j ? p1c : p0c;
        const int cc = j ? ch1 : ch0;
        const int iy = y0 + (p >> 4) + dy;
        const int ix = x0 + (p & 15) + dx;
        u32x4 val = {0u, 0u, 0u, 0u};
        if (iy >= 0 && iy < Hin && ix >= 0 && ix < Win)
          val = *(const u32x4*)(actb + ((size_t)iy * Win + ix) * IC + ic0 + cc * 8);
        *(u32x4*)(sB + p * LDS_STRIDE + cc * 8) = val;
      }
      if (wv == 0) __builtin_amdgcn_s_wait_tensorcnt((short)0);
      __syncthreads();

      const v16bf af = load_a_frag(sA, om * 16, lane);
#pragma unroll
      for (int i = 0; i < 4; ++i) {
        const v16bf bf = load_b_frag(sB, (pnb + i) * 16, lane);
        c[i] = __builtin_amdgcn_wmma_f32_16x16x32_bf16(false, af, false, bf,
                                                       (short)0, c[i], false, false);
      }
      __syncthreads();
    }
  }

  // epilogue: demod * acc + noise + bias -> lrelu(0.2)*sqrt(2)
  const int   hi   = lane >> 4;
  const int   colN = lane & 15;
  const float nsv  = nsPtr[0];
  const int   py   = phase >> 1, px = phase & 1;
  const int   ocb  = oc0 + om * 16 + hi * 8;      // 8 consecutive channels
  const float dmv[8] = {demod[b*256+ocb+0], demod[b*256+ocb+1], demod[b*256+ocb+2],
                        demod[b*256+ocb+3], demod[b*256+ocb+4], demod[b*256+ocb+5],
                        demod[b*256+ocb+6], demod[b*256+ocb+7]};
#pragma unroll
  for (int tile = 0; tile < 4; ++tile) {
    const int pl = (pnb + tile) * 16 + colN;
    const int y  = y0 + (pl >> 4);
    const int x  = x0 + (pl & 15);
    const int y2 = (MODE == 0) ? (2 * y + py) : y;
    const int x2 = (MODE == 0) ? (2 * x + px) : x;
    const float nz = nsv * noise[(size_t)b * 16384 + y2 * 128 + x2];
    float vv[8];
#pragma unroll
    for (int r = 0; r < 8; ++r) {
      float v = c[tile][r] * dmv[r] + nz + bias[ocb + r];
      vv[r] = (v > 0.f ? v : 0.2f * v) * SQRT2F;
    }
    u32x4 pk;
#pragma unroll
    for (int j = 0; j < 4; ++j)
      pk[j] = (unsigned int)f2bf(vv[2 * j]) | ((unsigned int)f2bf(vv[2 * j + 1]) << 16);
    const size_t hidx = ((size_t)b * 16384 + (size_t)y2 * 128 + x2) * 256 + ocb;
    *(u32x4*)(outH + hidx) = pk;                       // channel-last bf16
    if (MODE == 1) {
#pragma unroll
      for (int r = 0; r < 8; ++r)                      // h2 fp32, NCHW (d_out)
        outF[(((size_t)(b * 256 + ocb + r)) * 128 + y) * 128 + x] = vv[r];
    }
  }
}

// ---------------------------------------------------------------------------
// to_rgb (modulated 1x1, no demod) + bias2 + upfirdn upsampled skip of y.
// h2 is channel-last bf16 -> contiguous 512B per pixel, b128 reads.
// ---------------------------------------------------------------------------
__global__ void k_rgb(const unsigned short* __restrict__ h2, const float* __restrict__ s2,
                      const float* __restrict__ w2, const float* __restrict__ b2,
                      const float* __restrict__ y, float* __restrict__ outRGB)
{
  const float kv[4] = {1.f, 3.f, 3.f, 1.f};
  const int n = 8 * 128 * 128;
  for (int idx = blockIdx.x * blockDim.x + threadIdx.x; idx < n;
       idx += gridDim.x * blockDim.x) {
    const int b  = idx >> 14;
    const int oy = (idx >> 7) & 127;
    const int ox = idx & 127;

    const unsigned short* hp  = h2 + ((size_t)b * 16384 + oy * 128 + ox) * 256;
    const float*          s2b = s2 + b * 256;
    float a0 = 0.f, a1 = 0.f, a2 = 0.f;
    for (int c8 = 0; c8 < 256; c8 += 8) {
      const u32x4 pk = *(const u32x4*)(hp + c8);
#pragma unroll
      for (int j = 0; j < 4; ++j) {
        const float h0 = bf2f((unsigned short)(pk[j] & 0xFFFFu));
        const float h1 = bf2f((unsigned short)(pk[j] >> 16));
        const int   c  = c8 + 2 * j;
        const float hs0 = h0 * (s2b[c] * 0.0625f);       // sc2 = 1/16
        const float hs1 = h1 * (s2b[c + 1] * 0.0625f);
        a0 += hs0 * w2[c]       + hs1 * w2[c + 1];
        a1 += hs0 * w2[256 + c] + hs1 * w2[257 + c];
        a2 += hs0 * w2[512 + c] + hs1 * w2[513 + c];
      }
    }

    // y_up: depthwise [1,3,3,1]^2/16 upsample (2 taps per dim per parity)
    const int py = oy & 1, px = ox & 1, m = oy >> 1, nn = ox >> 1;
    float up[3] = {0.f, 0.f, 0.f};
    for (int uu = 0; uu < 2; ++uu) {
      const int iy = m + py + uu - 1;
      if (iy < 0 || iy >= 64) continue;
      const float wy = kv[py + 2 * uu];
      for (int vv = 0; vv < 2; ++vv) {
        const int ix = nn + px + vv - 1;
        if (ix < 0 || ix >= 64) continue;
        const float w = wy * kv[px + 2 * vv] * (1.f / 16.f);
        for (int o = 0; o < 3; ++o)
          up[o] += w * y[((size_t)(b * 3 + o) * 64 + iy) * 64 + ix];
      }
    }
    const float res[3] = {a0 + b2[0] + up[0], a1 + b2[1] + up[1], a2 + b2[2] + up[2]};
    for (int o = 0; o < 3; ++o)
      outRGB[(((size_t)(b * 3 + o)) * 128 + oy) * 128 + ox] = res[o];
  }
}

// ---------------------------------------------------------------------------
extern "C" void kernel_launch(void* const* d_in, const int* in_sizes, int n_in,
                              void* d_out, int out_size, void* d_ws, size_t ws_size,
                              hipStream_t stream)
{
  (void)in_sizes; (void)n_in; (void)out_size; (void)ws_size;

  const float* x      = (const float*)d_in[0];
  const float* dl     = (const float*)d_in[1];
  const float* yimg   = (const float*)d_in[2];
  const float* noise0 = (const float*)d_in[3];
  const float* noise1 = (const float*)d_in[4];
  const float* w0     = (const float*)d_in[5];
  const float* m0w    = (const float*)d_in[6];
  const float* m0b    = (const float*)d_in[7];
  const float* bias0  = (const float*)d_in[8];
  const float* ns0    = (const float*)d_in[9];
  const float* w1     = (const float*)d_in[10];
  const float* m1w    = (const float*)d_in[11];
  const float* m1b    = (const float*)d_in[12];
  const float* bias1  = (const float*)d_in[13];
  const float* ns1    = (const float*)d_in[14];
  const float* w2     = (const float*)d_in[15];
  const float* m2w    = (const float*)d_in[16];
  const float* m2b    = (const float*)d_in[17];
  const float* bias2  = (const float*)d_in[18];

  // workspace layout (~178.5 MB total)
  char* ws = (char*)d_ws;
  float*          s0  = (float*)(ws);                               // 16 KB
  float*          s1  = (float*)(ws + 16384);                       //  8 KB
  float*          s2  = (float*)(ws + 24576);                       //  8 KB
  float*          d0  = (float*)(ws + 32768);                       //  8 KB
  float*          d1  = (float*)(ws + 40960);                       //  8 KB
  unsigned short* Wp0 = (unsigned short*)(ws + 65536);              // 9.44 MB
  unsigned short* Wp1 = (unsigned short*)(ws + 65536 + 9437184);    // 1.18 MB
  unsigned short* xm  = (unsigned short*)(ws + 65536 + 9437184 + 1179648);             // 33.6 MB
  unsigned short* hb  = (unsigned short*)(ws + 65536 + 9437184 + 1179648 + 33554432);  // 67.1 MB
  unsigned short* h2b = (unsigned short*)(ws + 65536 + 9437184 + 1179648 + 33554432
                                             + 67108864);                              // 67.1 MB

  float* outH2  = (float*)d_out;
  float* outRGB = outH2 + (size_t)8 * 256 * 128 * 128;

  k_style<<<8, 256, 0, stream>>>(dl, m0w, m0b, m1w, m1b, m2w, m2b, w0, w1,
                                 s0, s1, s2, d0, d1);
  k_pack_w0<<<512, 256, 0, stream>>>(w0, Wp0);
  k_pack_w1<<<(256 * 2304 + 255) / 256, 256, 0, stream>>>(w1, Wp1);
  {
    dim3 g(128, 16, 8);   // (4096/32 px) x (512/32 ch) x batch
    k_modx<<<g, 256, 0, stream>>>(x, s0, xm);
  }

  // conv0_up + blur as 4 phase 3x3 convs on the 64x64 modulated input
  for (int ph = 0; ph < 4; ++ph) {
    dim3 g(4 * 8, 4, 8);  // (64/16 x)*(64/8 y) px tiles, 4 oc tiles, 8 batch
    k_conv3x3_wmma<0><<<g, 256, 0, stream>>>(xm, Wp0 + (size_t)ph * 256 * 4608,
                                             d0, noise0, bias0, ns0,
                                             nullptr, hb, 512, 64, 64, ph);
  }
  // conv1 on 128x128
  {
    dim3 g(8 * 16, 4, 8);
    k_conv3x3_wmma<1><<<g, 256, 0, stream>>>(hb, Wp1, d1, noise1, bias1, ns1,
                                             outH2, h2b, 256, 128, 128, 0);
  }
  k_rgb<<<512, 256, 0, stream>>>(h2b, s2, w2, bias2, yimg, outRGB);
}
